// Autograd4bitQuantLinear_45518063403104
// MI455X (gfx1250) — matrix operand order repaired and checked by reading, not compiled, not run on hardware
//
#include <hip/hip_runtime.h>

// CDNA5 / gfx1250 4-bit-quant linear: out = x @ dequant(qweight) + bias
//   x: (8192, 4096) f32   qweight: (512, 4096) i32 (8 nibbles along K)
//   scales/zeros: (4096,) f32   bias: (4096,) f32   out: (8192, 4096) f32
// Compute-bound at f16 WMMA rates -> dequant to f16, accumulate f32 with
// V_WMMA_F32_16X16X32_F16. 256x128 block tile (amortizes dequant 2x better),
// K-tile 32, double-buffered LDS, 8 waves of 64x64 (16 WMMA each per K-step).

typedef __attribute__((ext_vector_type(16))) _Float16 v16h;
typedef __attribute__((ext_vector_type(8)))  _Float16 v8h;
typedef __attribute__((ext_vector_type(2)))  _Float16 v2h;
typedef __attribute__((ext_vector_type(8)))  float    v8f;
typedef __attribute__((ext_vector_type(4)))  float    v4f;

#define M_TOT 8192
#define K_TOT 4096
#define N_TOT 4096
#define BM 256
#define BN 128
#define BK 32
#define KSTEPS (K_TOT / BK)   // 128
#define LDSS 40               // padded row stride in halfs (80B, 16B-aligned)

__global__ __launch_bounds__(256)
void q4_gemm_wmma_f16(const float* __restrict__ x,
                      const int*   __restrict__ qw,
                      const float* __restrict__ scales,
                      const float* __restrict__ zeros,
                      const float* __restrict__ bias,
                      float* __restrict__ out)
{
    __shared__ __align__(16) _Float16 lA[2][BM][LDSS];  // [m][k], k in [0,32)
    __shared__ __align__(16) _Float16 lB[2][BN][LDSS];  // [n][k], k in [0,32)

    const int t    = threadIdx.x;
    const int lane = t & 31;
    const int w    = t >> 5;          // wave id 0..7
    const int wm   = w >> 1;          // 0..3  (64 rows each)
    const int wn   = w & 1;           // 0..1  (64 cols each)

    const int mblk = blockIdx.y * BM;
    const int nblk = blockIdx.x * BN;

    // ---- staging assignments ----
    // A: thread t stages one full x row (32 floats): row t of the tile.
    // B: thread stages 2 packed dwords (16 int4): col bNL, packed rows {bPR, bPR+2}
    const int bNL  = t & 127;          // 0..127
    const int bPR  = t >> 7;           // 0..1

    // per-column dequant constants in packed f16:  w = n * s - z  (n exact)
    const _Float16 hs = (_Float16)scales[nblk + bNL];
    const _Float16 hz = (_Float16)zeros [nblk + bNL];
    const v2h vs   = {hs, hs};
    const v2h vz   = {hz, hz};
    const v2h voff = {(_Float16)1024.0f, (_Float16)1024.0f};

    const float* xrow = x + (size_t)(mblk + t) * K_TOT;

    // accumulators: 64x64 wave tile = 4x4 fragments of 16x16 f32
    v8f acc[4][4];
    #pragma unroll
    for (int i = 0; i < 4; ++i)
        #pragma unroll
        for (int j = 0; j < 4; ++j)
            acc[i][j] = (v8f){0.f,0.f,0.f,0.f,0.f,0.f,0.f,0.f};

    // global->register staging
    v4f fx[8];
    unsigned q0, q1;

    auto load_regs = [&](int kt) {
        const v4f* xp = (const v4f*)(xrow + kt * BK);
        #pragma unroll
        for (int j = 0; j < 8; ++j) fx[j] = xp[j];
        const int kp = kt * (BK / 8);  // 4 packed rows per K-tile
        q0 = (unsigned)qw[(size_t)(kp + bPR    ) * N_TOT + nblk + bNL];
        q1 = (unsigned)qw[(size_t)(kp + bPR + 2) * N_TOT + nblk + bNL];
    };

    // unpack 8 nibbles of one dword -> 8 dequantized f16 (0x6400|n trick:
    // bitcast gives 1024+n exactly; subtract 1024 (exact) then packed fma).
    auto dequant8 = [&](unsigned q) -> v8h {
        v8h r;
        #pragma unroll
        for (int p = 0; p < 4; ++p) {
            unsigned u = 0x64006400u
                       | ((q >> (8 * p)) & 0xFu)
                       | (((q >> (8 * p)) & 0xF0u) << 12);
            v2h h;
            __builtin_memcpy(&h, &u, 4);
            v2h wv = (h - voff) * vs - vz;   // v_pk_add_f16 + v_pk_fma_f16
            r[2 * p]     = wv[0];
            r[2 * p + 1] = wv[1];
        }
        return r;
    };

    auto store_lds = [&](int buf) {
        // A: f32 -> f16 (v_cvt_pk_f16_f32), four b128 stores (one x row)
        #pragma unroll
        for (int c = 0; c < 4; ++c) {
            v8h h;
            #pragma unroll
            for (int j = 0; j < 4; ++j) {
                h[j]     = (_Float16)fx[2 * c][j];
                h[j + 4] = (_Float16)fx[2 * c + 1][j];
            }
            *(v8h*)&lA[buf][t][8 * c] = h;
        }
        // B: two dwords -> 16 weights, two b128 stores
        *(v8h*)&lB[buf][bNL][bPR * 8]      = dequant8(q0);  // k = bPR*8 .. +8
        *(v8h*)&lB[buf][bNL][bPR * 8 + 16] = dequant8(q1);  // k = (bPR+2)*8
    };

    auto compute = [&](int buf) {
        const int l15  = lane & 15;
        const int aKlo = (lane >> 4) << 3;   // 0 or 8   (A: K split per half-wave)
        const int bKlo = (lane >> 4) << 3;   // 0 or 8   (B: K split per half-wave)
        v16h afr[4];
        #pragma unroll
        for (int fm = 0; fm < 4; ++fm) {
            const int row = wm * 64 + fm * 16 + l15;
            v8h lo = *(const v8h*)&lA[buf][row][aKlo];
            v8h hi = *(const v8h*)&lA[buf][row][aKlo + 16];
            afr[fm] = __builtin_shufflevector(lo, hi,
                0,1,2,3,4,5,6,7,8,9,10,11,12,13,14,15);
        }
        #pragma unroll
        for (int fn = 0; fn < 4; ++fn) {
            const int col = wn * 64 + fn * 16 + l15;
            v8h lo = *(const v8h*)&lB[buf][col][bKlo];
            v8h hi = *(const v8h*)&lB[buf][col][bKlo + 16];
            v16h bfr = __builtin_shufflevector(lo, hi,
                0,1,2,3,4,5,6,7,8,9,10,11,12,13,14,15);
            #pragma unroll
            for (int fm = 0; fm < 4; ++fm)
                acc[fm][fn] = __builtin_amdgcn_wmma_f32_16x16x32_f16(
                    false, afr[fm], false, bfr, (short)0, acc[fm][fn],
                    false, false);
        }
    };

    // ---- main pipeline: prefetch-to-regs / WMMA / store-other-buffer ----
    load_regs(0);
    store_lds(0);
    __syncthreads();
    int cur = 0;
    #pragma unroll 1
    for (int kt = 0; kt < KSTEPS; ++kt) {
        if (kt + 1 < KSTEPS) {
            load_regs(kt + 1);
            if (kt + 2 < KSTEPS) {  // global_prefetch_b8 two tiles ahead
                __builtin_prefetch(xrow + (kt + 2) * BK, 0, 1);
                __builtin_prefetch(
                    qw + (size_t)((kt + 2) * 4 + bPR) * N_TOT + nblk + bNL, 0, 1);
            }
        }
        compute(cur);
        if (kt + 1 < KSTEPS) {
            store_lds(cur ^ 1);
            __syncthreads();
            cur ^= 1;
        }
    }

    // ---- epilogue: C layout -> lane (l&15)=N, VGPR r + 8*(lane>>4) = M ----
    #pragma unroll
    for (int fn = 0; fn < 4; ++fn) {
        const int n  = nblk + wn * 64 + fn * 16 + (lane & 15);
        const float bv = bias[n];
        #pragma unroll
        for (int fm = 0; fm < 4; ++fm) {
            const int mbase = mblk + wm * 64 + fm * 16 + ((lane >> 4) << 3);
            #pragma unroll
            for (int r = 0; r < 8; ++r)
                out[(size_t)(mbase + r) * N_TOT + n] = acc[fm][fn][r] + bv;
        }
    }
}

extern "C" void kernel_launch(void* const* d_in, const int* in_sizes, int n_in,
                              void* d_out, int out_size, void* d_ws, size_t ws_size,
                              hipStream_t stream) {
    const float* x      = (const float*)d_in[0];
    const int*   qw     = (const int*)  d_in[1];
    const float* scales = (const float*)d_in[2];
    const float* zeros  = (const float*)d_in[3];
    const float* bias   = (const float*)d_in[4];
    float*       out    = (float*)d_out;

    dim3 grid(N_TOT / BN, M_TOT / BM);  // (32, 32); x-major => L2 reuse of x rows
    q4_gemm_wmma_f16<<<grid, 256, 0, stream>>>(x, qw, scales, zeros, bias, out);
}
